// PhraseModel_45535243272917
// MI455X (gfx1250) — matrix operand order repaired and checked by reading, not compiled, not run on hardware
//
#include <hip/hip_runtime.h>

// ---------------------------------------------------------------------------
// VQ nearest-code lookup, MI455X (gfx1250, wave32, WMMA bf16 16x16x32)
//   dist2[n,k] = ||c_k||^2 - 2 * <z_n, c_k>   (+ ||z_n||^2, row-constant)
//   out[n,:]   = codebook[argmin_k dist2]     (exact fp32 gather)
// 32 rows/wave (2 A frags share each B frag), pipelined B, prefetched A.
// ---------------------------------------------------------------------------

typedef __attribute__((ext_vector_type(16))) __bf16 v16bf;
typedef __attribute__((ext_vector_type(8)))  __bf16 bf16x8;
typedef __attribute__((ext_vector_type(8)))  float  v8f;

#define D_DIM          510
#define K_CODES        128
#define CB_PITCH       520   // bf16 elems per LDS row: conflict-free stride, 16B aligned
#define ROWS_PER_BLOCK 256   // 8 waves x 32 rows
#define KCHUNKS        16    // K padded 510 -> 512

// ---- helpers --------------------------------------------------------------

// A-matrix raw fetch: 16 fp32 per lane per chunk as 8 x float2 (8B aligned).
// Mapping to fragment elems: j=0..7 <- r[0..3].{x,y} ; j=8..15 <- r[4..7].{x,y}
__device__ __forceinline__ void load_a_raw(const float* __restrict__ zr, int kc,
                                           int half, float2 r[8]) {
    const int k0 = kc * 32 + half * 8;
    if (kc < KCHUNKS - 1) {
#pragma unroll
        for (int p = 0; p < 4; ++p) {
            r[p]     = *(const float2*)(zr + k0 + 2 * p);
            r[4 + p] = *(const float2*)(zr + k0 + 16 + 2 * p);
        }
    } else {  // tail chunk: K 480..511, zero-pad k >= 510
#pragma unroll
        for (int p = 0; p < 4; ++p) {
            const int ka = k0 + 2 * p, kb = k0 + 16 + 2 * p;
            float2 t, u;
            t.x = (ka     < D_DIM) ? zr[ka]     : 0.0f;
            t.y = (ka + 1 < D_DIM) ? zr[ka + 1] : 0.0f;
            u.x = (kb     < D_DIM) ? zr[kb]     : 0.0f;
            u.y = (kb + 1 < D_DIM) ? zr[kb + 1] : 0.0f;
            r[p]     = t;
            r[4 + p] = u;
        }
    }
}

// fp32 -> bf16 fragment (native fptrunc; RNE)
__device__ __forceinline__ v16bf cvt_a(const float2 r[8]) {
    v16bf a;
#pragma unroll
    for (int p = 0; p < 4; ++p) {
        a[2 * p]         = (__bf16)r[p].x;
        a[2 * p + 1]     = (__bf16)r[p].y;
        a[8 + 2 * p]     = (__bf16)r[4 + p].x;
        a[8 + 2 * p + 1] = (__bf16)r[4 + p].y;
    }
    return a;
}

// B fragment (32x16): lane = column code, k = 8*half+0..7 and 16+8*half+0..7 (two ds_load_b128)
__device__ __forceinline__ v16bf load_b(const __bf16* sCB, int nt, int kc,
                                        int half, int l16) {
    const __bf16* bp = sCB + (size_t)(nt * 16 + l16) * CB_PITCH + kc * 32 + half * 8;
    bf16x8 lo = *(const bf16x8*)bp;
    bf16x8 hi = *(const bf16x8*)(bp + 16);
    v16bf b;
#pragma unroll
    for (int i = 0; i < 8; ++i) { b[i] = lo[i]; b[i + 8] = hi[i]; }
    return b;
}

// argmin over 128 codes for one 16-row accumulator bank.
// C layout: VGPR r holds row 8*half + r, col n = nt*16 + l16.
__device__ __forceinline__ void argmin16(const v8f* acc, const float* sCsq,
                                         int l16, int half,
                                         int idxLow[8], int idxHigh[8]) {
    int rowBest[8];
#pragma unroll
    for (int r = 0; r < 8; ++r) {
        float bd = 3.4e38f;
        int   bn = 0;
#pragma unroll
        for (int nt = 0; nt < 8; ++nt) {
            const int   n  = nt * 16 + l16;
            const float d2 = sCsq[n] - 2.0f * acc[nt][r];
            if (d2 < bd || (d2 == bd && n < bn)) { bd = d2; bn = n; }
        }
#pragma unroll
        for (int mk = 1; mk < 16; mk <<= 1) {
            const float od = __shfl_xor(bd, mk, 32);
            const int   on = __shfl_xor(bn, mk, 32);
            if (od < bd || (od == bd && on < bn)) { bd = od; bn = on; }
        }
        rowBest[r] = bn;
    }
#pragma unroll
    for (int r = 0; r < 8; ++r) {
        const int other = __shfl_xor(rowBest[r], 16, 32);
        idxLow[r]  = (half == 0) ? rowBest[r] : other;   // rows base + r
        idxHigh[r] = (half == 0) ? other : rowBest[r];   // rows base + 8 + r
    }
}

// ---- kernel ---------------------------------------------------------------

extern "C" __global__ __launch_bounds__(256)
void vq_wmma_bf16_kernel(const float* __restrict__ z,
                         const float* __restrict__ cb,
                         float* __restrict__ out)
{
    extern __shared__ char smem_raw[];
    __bf16* sCB  = (__bf16*)smem_raw;                                   // [128][520]
    float*  sCsq = (float*)(smem_raw + (size_t)K_CODES * CB_PITCH * 2); // [128]

    const int tid = threadIdx.x;

    // Stage codebook into LDS as bf16 (zero-padded K >= 510)
    for (int idx = tid; idx < K_CODES * CB_PITCH; idx += 256) {
        const int n = idx / CB_PITCH;
        const int k = idx - n * CB_PITCH;
        sCB[idx] = (__bf16)((k < D_DIM) ? cb[n * D_DIM + k] : 0.0f);
    }
    // ||c_k||^2 in exact fp32
    if (tid < K_CODES) {
        const float* c = cb + tid * D_DIM;
        float s = 0.0f;
        for (int d = 0; d < D_DIM; ++d) s = fmaf(c[d], c[d], s);
        sCsq[tid] = s;
    }
    __syncthreads();

    const int wave = tid >> 5;
    const int lane = tid & 31;
    const int half = lane >> 4;
    const int l16  = lane & 15;
    const int m0   = blockIdx.x * ROWS_PER_BLOCK + wave * 32;

    v8f acc0[8], acc1[8];
    const v8f zero8 = {0.f, 0.f, 0.f, 0.f, 0.f, 0.f, 0.f, 0.f};
#pragma unroll
    for (int i = 0; i < 8; ++i) { acc0[i] = zero8; acc1[i] = zero8; }

    const float* zr0 = z + (size_t)(m0 + l16) * D_DIM;        // rows m0   .. m0+15
    const float* zr1 = z + (size_t)(m0 + 16 + l16) * D_DIM;   // rows m0+16.. m0+31

    // prime A prefetch
    float2 r0[8], r1[8];
    load_a_raw(zr0, 0, half, r0);
    load_a_raw(zr1, 0, half, r1);

    for (int kc = 0; kc < KCHUNKS; ++kc) {
        const v16bf a0 = cvt_a(r0);
        const v16bf a1 = cvt_a(r1);
        if (kc + 1 < KCHUNKS) {          // prefetch next chunk while WMMAs run
            load_a_raw(zr0, kc + 1, half, r0);
            load_a_raw(zr1, kc + 1, half, r1);
        }
        // 2-deep pipelined B; each B fragment feeds two WMMAs (acc0/acc1)
        v16bf bcur = load_b(sCB, 0, kc, half, l16);
#pragma unroll
        for (int nt = 0; nt < 8; ++nt) {
            v16bf bnext = bcur;
            if (nt + 1 < 8) bnext = load_b(sCB, nt + 1, kc, half, l16);
            acc0[nt] = __builtin_amdgcn_wmma_f32_16x16x32_bf16(
                false, a0, false, bcur, (short)0, acc0[nt], false, false);
            acc1[nt] = __builtin_amdgcn_wmma_f32_16x16x32_bf16(
                false, a1, false, bcur, (short)0, acc1[nt], false, false);
            bcur = bnext;
        }
    }

    // per-row argmin for both 16-row banks
    int loA[8], hiA[8], loB[8], hiB[8];
    argmin16(acc0, sCsq, l16, half, loA, hiA);
    argmin16(acc1, sCsq, l16, half, loB, hiB);

    // exact fp32 gather of winning codebook rows (coalesced b32 stores)
#pragma unroll
    for (int r = 0; r < 8; ++r) {
        const float* src = cb + (size_t)loA[r] * D_DIM;
        float*       dst = out + (size_t)(m0 + r) * D_DIM;
        for (int d = lane; d < D_DIM; d += 32) dst[d] = src[d];
    }
#pragma unroll
    for (int r = 0; r < 8; ++r) {
        const float* src = cb + (size_t)hiA[r] * D_DIM;
        float*       dst = out + (size_t)(m0 + 8 + r) * D_DIM;
        for (int d = lane; d < D_DIM; d += 32) dst[d] = src[d];
    }
#pragma unroll
    for (int r = 0; r < 8; ++r) {
        const float* src = cb + (size_t)loB[r] * D_DIM;
        float*       dst = out + (size_t)(m0 + 16 + r) * D_DIM;
        for (int d = lane; d < D_DIM; d += 32) dst[d] = src[d];
    }
#pragma unroll
    for (int r = 0; r < 8; ++r) {
        const float* src = cb + (size_t)hiB[r] * D_DIM;
        float*       dst = out + (size_t)(m0 + 24 + r) * D_DIM;
        for (int d = lane; d < D_DIM; d += 32) dst[d] = src[d];
    }
}

extern "C" void kernel_launch(void* const* d_in, const int* in_sizes, int n_in,
                              void* d_out, int out_size, void* d_ws, size_t ws_size,
                              hipStream_t stream)
{
    const float* z  = (const float*)d_in[0];   // [N, 510] fp32
    const float* cb = (const float*)d_in[1];   // [128, 510] fp32
    float* out = (float*)d_out;                // [N, 510] fp32

    const int N    = in_sizes[0] / D_DIM;      // 131072
    const int grid = N / ROWS_PER_BLOCK;       // 512 blocks
    const size_t shmem = (size_t)K_CODES * CB_PITCH * 2 + K_CODES * sizeof(float);

    hipLaunchKernelGGL(vq_wmma_bf16_kernel, dim3(grid), dim3(256), shmem, stream,
                       z, cb, out);
}
